// MultiHeadAttention_29703993819975
// MI455X (gfx1250) — matrix operand order repaired and checked by reading, compiled, not run on hardware
//
#include <hip/hip_runtime.h>
#include <hip/hip_bf16.h>
#include <math.h>

// ---------------------------------------------------------------------------
// MHA forward for B=2,S=2048,D=1024,H=16,HD=64 on gfx1250 (MI455X).
// All GEMMs use v_wmma_f32_16x16x32_bf16 (bf16 inputs, f32 accumulate).
// Softmax kept in f32.  attn [B,H,S,S] is materialized directly in d_out
// (required by the reference) => bound by ~2.1 GB of attn traffic
// (~92us at 23.3 TB/s).  Global staging: 128-bit loads everywhere; the
// scores kernel stages its (pure-copy) bf16 tiles with
// global_load_async_to_lds_b128 (ASYNCcnt path, no VGPR round trip).
// LDS pitch is 40 halfwords everywhere: 16B-aligned rows => fragment reads
// collapse to ds_load_b128, and the 20-bank row stride is conflict-free
// across 16 lanes (5 coprime to 16 => 16 disjoint 4-bank groups).
// ---------------------------------------------------------------------------

typedef __attribute__((ext_vector_type(16))) __bf16 v16bf;
typedef __attribute__((ext_vector_type(8)))  float  v8f;

#define B_   2
#define S_   2048
#define D_   1024
#define H_   16
#define HD_  64

__device__ __forceinline__ unsigned short f2bf(float f) {
    // round-to-nearest-even f32 -> bf16
    unsigned int u = __float_as_uint(f);
    u += 0x7fffu + ((u >> 16) & 1u);
    return (unsigned short)(u >> 16);
}

__device__ __forceinline__ unsigned int pack2bf(float a, float b) {
    return (unsigned int)f2bf(a) | ((unsigned int)f2bf(b) << 16);
}

union Frag {
    unsigned int u[8];
    v16bf v;
};

// 32-bit LDS byte offset of a __shared__-derived pointer.
__device__ __forceinline__ unsigned lds_off_of(const void* p) {
    return (unsigned)(unsigned long long)(
        (const __attribute__((address_space(3))) char*)p);
}

// Async copy 16B from global directly into LDS (tracked by ASYNCcnt).
__device__ __forceinline__ void async_copy_b128(unsigned lds_off, const void* gsrc) {
    asm volatile("global_load_async_to_lds_b128 %0, %1, off"
                 :: "v"(lds_off), "v"(gsrc)
                 : "memory");
}

__device__ __forceinline__ void wait_async0() {
    asm volatile("s_wait_asynccnt 0" ::: "memory");
}

// ---------------------------------------------------------------------------
// Generic tiled GEMM:  C[M,N] = A[M,K] * W[K,N] * scale + bias[N]
//   A:  f32 or bf16(ushort) row-major     W: f32 row-major [K,N]
//   C:  f32 or bf16(ushort) row-major
// Block tile 128x128, K-step 32, 8 waves (2 in M x 4 in N), wave tile 64x32.
// ---------------------------------------------------------------------------
template <bool A_BF16, bool OUT_BF16>
__global__ __launch_bounds__(256) void gemm_bias_wmma(
    const void* __restrict__ Ap, const float* __restrict__ W,
    const float* __restrict__ bias, void* __restrict__ Cp,
    int M, int N, int K, float scale)
{
    constexpr int BM = 128, BN = 128, BK = 32, LK = 40;
    __shared__ unsigned short As[BM * LK];
    __shared__ unsigned short Bs[BN * LK];  // stored transposed: Bs[n][k]

    const int tid   = threadIdx.x;
    const int m0    = blockIdx.y * BM;
    const int n0    = blockIdx.x * BN;
    const int wave  = tid >> 5;
    const int lane  = tid & 31;
    const int wm    = (wave & 1) * 64;   // wave origin in M
    const int wn    = (wave >> 1) * 32;  // wave origin in N
    const int lrow  = lane & 15;
    const int lhalf = lane >> 4;

    const float*          Af = (const float*)Ap;
    const unsigned short* Ab = (const unsigned short*)Ap;

    v8f zero = {0.f, 0.f, 0.f, 0.f, 0.f, 0.f, 0.f, 0.f};
    v8f acc[4][2];
    #pragma unroll
    for (int i = 0; i < 4; ++i)
        #pragma unroll
        for (int j = 0; j < 2; ++j) acc[i][j] = zero;

    for (int k0 = 0; k0 < K; k0 += BK) {
        const bool more = (k0 + BK) < K;
        // ---- stage A tile 128x32 with 128-bit loads ----
        if constexpr (A_BF16) {
            #pragma unroll
            for (int i = 0; i < (BM * BK) / (256 * 8); ++i) {  // 2
                int base = (i * 256 + tid) * 8;
                int m = base >> 5, k = base & 31;
                const unsigned short* p = &Ab[(size_t)(m0 + m) * K + k0 + k];
                uint4 u = *(const uint4*)p;
                if (more) __builtin_prefetch(p + BK, 0, 1);
                *(unsigned int*)&As[m * LK + k + 0] = u.x;
                *(unsigned int*)&As[m * LK + k + 2] = u.y;
                *(unsigned int*)&As[m * LK + k + 4] = u.z;
                *(unsigned int*)&As[m * LK + k + 6] = u.w;
            }
        } else {
            #pragma unroll
            for (int i = 0; i < (BM * BK) / (256 * 4); ++i) {  // 4
                int base = (i * 256 + tid) * 4;
                int m = base >> 5, k = base & 31;
                const float* p = &Af[(size_t)(m0 + m) * K + k0 + k];
                float4 f = *(const float4*)p;
                if (more) __builtin_prefetch(p + BK, 0, 1);
                *(unsigned int*)&As[m * LK + k + 0] = pack2bf(f.x, f.y);
                *(unsigned int*)&As[m * LK + k + 2] = pack2bf(f.z, f.w);
            }
        }
        // ---- stage B tile transposed: Bs[n][k] = W[k0+k][n0+n] ----
        #pragma unroll
        for (int i = 0; i < (BN * BK) / (256 * 4); ++i) {      // 4
            int base = (i * 256 + tid) * 4;
            int k = base >> 7, n = base & 127;
            const float* p = &W[(size_t)(k0 + k) * N + n0 + n];
            float4 f = *(const float4*)p;
            if (more) __builtin_prefetch(p + (size_t)BK * N, 0, 1);
            Bs[(n + 0) * LK + k] = f2bf(f.x);
            Bs[(n + 1) * LK + k] = f2bf(f.y);
            Bs[(n + 2) * LK + k] = f2bf(f.z);
            Bs[(n + 3) * LK + k] = f2bf(f.w);
        }
        __syncthreads();

        Frag a[4], b[2];
        #pragma unroll
        for (int mi = 0; mi < 4; ++mi)
            #pragma unroll
            for (int v = 0; v < 8; ++v) {
                int kb = lhalf * 16 + v * 2;
                a[mi].u[v] = *(const unsigned int*)&As[(wm + mi * 16 + lrow) * LK + kb];
            }
        #pragma unroll
        for (int ni = 0; ni < 2; ++ni)
            #pragma unroll
            for (int v = 0; v < 8; ++v) {
                int kb = lhalf * 16 + v * 2;
                b[ni].u[v] = *(const unsigned int*)&Bs[(wn + ni * 16 + lrow) * LK + kb];
            }
        #pragma unroll
        for (int mi = 0; mi < 4; ++mi)
            #pragma unroll
            for (int ni = 0; ni < 2; ++ni)
                acc[mi][ni] = __builtin_amdgcn_wmma_f32_16x16x32_bf16(
                    false, a[mi].v, false, b[ni].v, (short)0, acc[mi][ni],
                    false, false);
        __syncthreads();
    }

    // epilogue: C layout lanes0-15: N=lane, M=r; lanes16-31: N=lane-16, M=r+8
    const int nl   = lane & 15;
    const int madd = (lane >> 4) * 8;
    #pragma unroll
    for (int mi = 0; mi < 4; ++mi)
        #pragma unroll
        for (int ni = 0; ni < 2; ++ni)
            #pragma unroll
            for (int r = 0; r < 8; ++r) {
                int gm = m0 + wm + mi * 16 + madd + r;
                int gn = n0 + wn + ni * 16 + nl;
                float val = acc[mi][ni][r] * scale + bias[gn];
                if constexpr (OUT_BF16)
                    ((unsigned short*)Cp)[(size_t)gm * N + gn] = f2bf(val);
                else
                    ((float*)Cp)[(size_t)gm * N + gn] = val;
            }
}

// ---------------------------------------------------------------------------
// scores[b,h,q,k] = scale * sum_d Q[b,q,h*64+d] * K[b,k,h*64+d]
// Q,K are bf16 in [B,S,D] layout.  Tiles are pure row-major copies, so they
// are staged with global_load_async_to_lds_b128 (ASYNCcnt, no VGPR bounce).
// ---------------------------------------------------------------------------
__global__ __launch_bounds__(256) void attn_scores_wmma(
    const unsigned short* __restrict__ Q,
    const unsigned short* __restrict__ Kmat,
    float* __restrict__ attn, float scale)
{
    constexpr int BM = 128, BN = 128, BK = 32, LK = 40;
    __shared__ unsigned short As[BM * LK];
    __shared__ unsigned short Bs[BN * LK];

    const int tid = threadIdx.x;
    const int bh  = blockIdx.z;
    const int b   = bh / H_;
    const int h   = bh % H_;
    const unsigned short* Qh = Q    + (size_t)b * S_ * D_ + h * HD_;
    const unsigned short* Kh = Kmat + (size_t)b * S_ * D_ + h * HD_;
    float* Out = attn + (size_t)bh * S_ * S_;

    const unsigned as_off = lds_off_of(As);
    const unsigned bs_off = lds_off_of(Bs);

    const int m0 = blockIdx.y * BM;  // query rows
    const int n0 = blockIdx.x * BN;  // key cols
    const int wave = tid >> 5, lane = tid & 31;
    const int wm = (wave & 1) * 64, wn = (wave >> 1) * 32;
    const int lrow = lane & 15, lhalf = lane >> 4;

    v8f zero = {0.f, 0.f, 0.f, 0.f, 0.f, 0.f, 0.f, 0.f};
    v8f acc[4][2];
    #pragma unroll
    for (int i = 0; i < 4; ++i)
        #pragma unroll
        for (int j = 0; j < 2; ++j) acc[i][j] = zero;

    #pragma unroll
    for (int k0 = 0; k0 < HD_; k0 += BK) {  // 2 iterations
        // ---- async-stage A: As[m][k] = Q[m0+m][k0+k], 16B chunks ----
        #pragma unroll
        for (int i = 0; i < 2; ++i) {
            int base = (i * 256 + tid) * 8;
            int m = base >> 5, k = base & 31;
            async_copy_b128(as_off + (unsigned)(m * LK + k) * 2,
                            &Qh[(size_t)(m0 + m) * D_ + k0 + k]);
        }
        // ---- async-stage B: Bs[n][d] = K[n0+n][d] (row-major) ----
        #pragma unroll
        for (int i = 0; i < 2; ++i) {
            int base = (i * 256 + tid) * 8;
            int n = base >> 5, k = base & 31;
            async_copy_b128(bs_off + (unsigned)(n * LK + k) * 2,
                            &Kh[(size_t)(n0 + n) * D_ + k0 + k]);
        }
        wait_async0();
        __syncthreads();

        Frag a[4], bf[2];
        #pragma unroll
        for (int mi = 0; mi < 4; ++mi)
            #pragma unroll
            for (int v = 0; v < 8; ++v) {
                int kb = lhalf * 16 + v * 2;
                a[mi].u[v] = *(const unsigned int*)&As[(wm + mi * 16 + lrow) * LK + kb];
            }
        #pragma unroll
        for (int ni = 0; ni < 2; ++ni)
            #pragma unroll
            for (int v = 0; v < 8; ++v) {
                int kb = lhalf * 16 + v * 2;
                bf[ni].u[v] = *(const unsigned int*)&Bs[(wn + ni * 16 + lrow) * LK + kb];
            }
        #pragma unroll
        for (int mi = 0; mi < 4; ++mi)
            #pragma unroll
            for (int ni = 0; ni < 2; ++ni)
                acc[mi][ni] = __builtin_amdgcn_wmma_f32_16x16x32_bf16(
                    false, a[mi].v, false, bf[ni].v, (short)0, acc[mi][ni],
                    false, false);
        __syncthreads();
    }

    const int nl = lane & 15, madd = (lane >> 4) * 8;
    #pragma unroll
    for (int mi = 0; mi < 4; ++mi)
        #pragma unroll
        for (int ni = 0; ni < 2; ++ni)
            #pragma unroll
            for (int r = 0; r < 8; ++r) {
                int gm = m0 + wm + mi * 16 + madd + r;
                int gn = n0 + wn + ni * 16 + nl;
                Out[(size_t)gm * S_ + gn] = acc[mi][ni][r] * scale;
            }
}

// ---------------------------------------------------------------------------
// In-place row softmax over the last axis (S_=2048), one block per row.
// ---------------------------------------------------------------------------
__global__ __launch_bounds__(256) void softmax_rows(float* __restrict__ attn)
{
    __shared__ float red[256];
    const int tid = threadIdx.x;
    float* p = attn + (size_t)blockIdx.x * S_;

    float v[8];
    float mx = -3.402823466e+38f;
    #pragma unroll
    for (int i = 0; i < 8; ++i) {
        v[i] = p[tid + i * 256];
        mx = fmaxf(mx, v[i]);
    }
    red[tid] = mx;
    __syncthreads();
    for (int s = 128; s > 0; s >>= 1) {
        if (tid < s) red[tid] = fmaxf(red[tid], red[tid + s]);
        __syncthreads();
    }
    mx = red[0];
    __syncthreads();

    float sum = 0.f;
    #pragma unroll
    for (int i = 0; i < 8; ++i) {
        v[i] = __expf(v[i] - mx);
        sum += v[i];
    }
    red[tid] = sum;
    __syncthreads();
    for (int s = 128; s > 0; s >>= 1) {
        if (tid < s) red[tid] += red[tid + s];
        __syncthreads();
    }
    const float inv = 1.f / red[0];
    #pragma unroll
    for (int i = 0; i < 8; ++i) p[tid + i * 256] = v[i] * inv;
}

// ---------------------------------------------------------------------------
// O[b,q,h*64+d] = sum_k attn[b,h,q,k] * V[b,k,h*64+d]
// attn f32 (converted to bf16 on stage), V bf16, O bf16.
// Block tile 128x64, K-step 32, 8 waves (2 in M x 4 in N), wave tile 64x16.
// ---------------------------------------------------------------------------
__global__ __launch_bounds__(256) void attn_v_wmma(
    const float* __restrict__ attn,
    const unsigned short* __restrict__ V,
    unsigned short* __restrict__ O)
{
    constexpr int BM = 128, BK = 32, LK = 40;
    __shared__ unsigned short As[BM * LK];
    __shared__ unsigned short Bs[64 * LK];  // Bs[d][s]

    const int tid = threadIdx.x;
    const int bh  = blockIdx.z;
    const int b   = bh / H_;
    const int h   = bh % H_;
    const float*          Ah = attn + (size_t)bh * S_ * S_;
    const unsigned short* Vh = V + (size_t)b * S_ * D_ + h * HD_;
    unsigned short*       Oh = O + (size_t)b * S_ * D_ + h * HD_;

    const int m0 = blockIdx.x * BM;
    const int wave = tid >> 5, lane = tid & 31;
    const int wm = (wave & 1) * 64;     // 2 waves in M
    const int wn = (wave >> 1) * 16;    // 4 waves in N (64 cols)
    const int lrow = lane & 15, lhalf = lane >> 4;

    v8f zero = {0.f, 0.f, 0.f, 0.f, 0.f, 0.f, 0.f, 0.f};
    v8f acc[4];
    #pragma unroll
    for (int i = 0; i < 4; ++i) acc[i] = zero;

    for (int k0 = 0; k0 < S_; k0 += BK) {
        const bool more = (k0 + BK) < S_;
        // ---- A: 128x32 from attn f32 (128-bit loads) ----
        #pragma unroll
        for (int i = 0; i < 4; ++i) {
            int base = (i * 256 + tid) * 4;
            int m = base >> 5, k = base & 31;
            const float* p = &Ah[(size_t)(m0 + m) * S_ + k0 + k];
            float4 f = *(const float4*)p;
            if (more) __builtin_prefetch(p + BK, 0, 1);
            *(unsigned int*)&As[m * LK + k + 0] = pack2bf(f.x, f.y);
            *(unsigned int*)&As[m * LK + k + 2] = pack2bf(f.z, f.w);
        }
        // ---- B: Bs[d][s] = V[k0+s][h*64+d]  (coalesced 128-bit reads) ----
        {
            int base = tid * 8;               // 2048 halfwords, one pass
            int s = base >> 6, d = base & 63;
            const unsigned short* p = &Vh[(size_t)(k0 + s) * D_ + d];
            uint4 u = *(const uint4*)p;
            if (more) __builtin_prefetch(p + (size_t)BK * D_, 0, 1);
            unsigned int w[4] = {u.x, u.y, u.z, u.w};
            #pragma unroll
            for (int j = 0; j < 4; ++j) {
                Bs[(d + 2 * j + 0) * LK + s] = (unsigned short)(w[j] & 0xffffu);
                Bs[(d + 2 * j + 1) * LK + s] = (unsigned short)(w[j] >> 16);
            }
        }
        __syncthreads();

        Frag a[4], bf;
        #pragma unroll
        for (int mi = 0; mi < 4; ++mi)
            #pragma unroll
            for (int v = 0; v < 8; ++v) {
                int kb = lhalf * 16 + v * 2;
                a[mi].u[v] = *(const unsigned int*)&As[(wm + mi * 16 + lrow) * LK + kb];
            }
        #pragma unroll
        for (int v = 0; v < 8; ++v) {
            int kb = lhalf * 16 + v * 2;
            bf.u[v] = *(const unsigned int*)&Bs[(wn + lrow) * LK + kb];
        }
        #pragma unroll
        for (int mi = 0; mi < 4; ++mi)
            acc[mi] = __builtin_amdgcn_wmma_f32_16x16x32_bf16(
                false, a[mi].v, false, bf.v, (short)0, acc[mi], false, false);
        __syncthreads();
    }

    const int nl = lane & 15, madd = (lane >> 4) * 8;
    #pragma unroll
    for (int mi = 0; mi < 4; ++mi)
        #pragma unroll
        for (int r = 0; r < 8; ++r) {
            int gm = m0 + wm + mi * 16 + madd + r;
            int gn = wn + nl;
            Oh[(size_t)gm * D_ + gn] = f2bf(acc[mi][r]);
        }
}

// ---------------------------------------------------------------------------
extern "C" void kernel_launch(void* const* d_in, const int* in_sizes, int n_in,
                              void* d_out, int out_size, void* d_ws, size_t ws_size,
                              hipStream_t stream)
{
    (void)in_sizes; (void)n_in; (void)out_size; (void)ws_size;

    const float* q_in = (const float*)d_in[0];
    const float* k_in = (const float*)d_in[1];
    const float* v_in = (const float*)d_in[2];
    const float* Wq   = (const float*)d_in[3];
    const float* bq   = (const float*)d_in[4];
    const float* Wk   = (const float*)d_in[5];
    const float* bk   = (const float*)d_in[6];
    const float* Wv   = (const float*)d_in[7];
    const float* bv   = (const float*)d_in[8];
    const float* Wo   = (const float*)d_in[9];
    const float* bo   = (const float*)d_in[10];

    const int M = B_ * S_;  // 4096

    // workspace: Q,K,V,O as bf16 [B,S,D]  (4 * 8 MB = 32 MB)
    unsigned short* Qb = (unsigned short*)d_ws;
    unsigned short* Kb = Qb + (size_t)M * D_;
    unsigned short* Vb = Kb + (size_t)M * D_;
    unsigned short* Ob = Vb + (size_t)M * D_;

    float* out_f = (float*)d_out;                    // [B,S,D]
    float* attn  = out_f + (size_t)M * D_;           // [B,H,S,S]

    dim3 blk(256);

    // 1) Q/K/V projections: f32 in, bf16 out
    dim3 gproj(D_ / 128, M / 128);
    gemm_bias_wmma<false, true><<<gproj, blk, 0, stream>>>(q_in, Wq, bq, Qb, M, D_, D_, 1.0f);
    gemm_bias_wmma<false, true><<<gproj, blk, 0, stream>>>(k_in, Wk, bk, Kb, M, D_, D_, 1.0f);
    gemm_bias_wmma<false, true><<<gproj, blk, 0, stream>>>(v_in, Wv, bv, Vb, M, D_, D_, 1.0f);

    // 2) scaled scores -> attn region of d_out (async-LDS staged)
    dim3 gsc(S_ / 128, S_ / 128, B_ * H_);
    attn_scores_wmma<<<gsc, blk, 0, stream>>>(Qb, Kb, attn, 0.125f);  // 1/sqrt(64)

    // 3) softmax in place (one block per row; B*H*S = 65536 rows)
    softmax_rows<<<dim3(B_ * H_ * S_), blk, 0, stream>>>(attn);

    // 4) attn @ V -> O (bf16, heads merged back to [B,S,D])
    dim3 gav(S_ / 128, 1, B_ * H_);
    attn_v_wmma<<<gav, blk, 0, stream>>>(attn, Vb, Ob);

    // 5) output projection: bf16 in, f32 out
    gemm_bias_wmma<true, false><<<gproj, blk, 0, stream>>>(Ob, Wo, bo, out_f, M, D_, D_, 1.0f);
}
